// EGCLayer_5214090297740
// MI455X (gfx1250) — matrix operand order repaired and checked by reading, not compiled, not run on hardware
//
#include <hip/hip_runtime.h>
#include <math.h>

#define NN 50000
#define EE 800000
#define HH 64

typedef __attribute__((ext_vector_type(16))) __bf16 v16bf;
typedef __attribute__((ext_vector_type(8)))  float  v8f;

// float -> bf16 via the native convert (compiler selects v_cvt_*bf16_f32 on gfx1250)
__device__ __forceinline__ unsigned short f2bf(float f) {
  union { __bf16 h; unsigned short u; } cv;
  cv.h = (__bf16)f;
  return cv.u;
}
__device__ __forceinline__ unsigned int pk2(float a, float b) {
  union { __bf16 h[2]; unsigned int u; } cv;
  cv.h[0] = (__bf16)a;
  cv.h[1] = (__bf16)b;
  return cv.u;
}
// Build a 16-element bf16 fragment from two 16-byte LDS chunks (ds_load_b128 x2)
__device__ __forceinline__ v16bf ld_frag(const unsigned short* p0, const unsigned short* p1) {
  union { uint4 q[2]; v16bf v; } u;
  u.q[0] = *reinterpret_cast<const uint4*>(p0);
  u.q[1] = *reinterpret_cast<const uint4*>(p1);
  return u.v;
}
#define WMMA_BF16(a, b, c) \
  __builtin_amdgcn_wmma_f32_16x16x32_bf16(false, (a), false, (b), (short)0, (c), false, false)

// ---------------------------------------------------------------------------
// Edge kernel: one 32-edge tile (two 16-row WMMA M-tiles) per wave.
// Each B (weight) fragment loaded once from LDS feeds two independent WMMAs.
//   m = relu(relu([feat_s|feat_d|sqd] @ W1 + b1) @ W2 + b2)
//   gate = sigmoid(m @ we + be);  atomically scatter m*gate into msum[dst]
// ---------------------------------------------------------------------------
__global__ __launch_bounds__(64) void egc_edge_kernel(
    const float* __restrict__ feat, const float* __restrict__ x,
    const int* __restrict__ src, const int* __restrict__ dst,
    const float* __restrict__ W1, const float* __restrict__ b1,
    const float* __restrict__ W2, const float* __restrict__ b2,
    const float* __restrict__ we, const float* __restrict__ be,
    float* __restrict__ msum)
{
  __shared__ __align__(16) unsigned short sW1T[64 * 128];  // W1^T (K padded to 128), bf16
  __shared__ __align__(16) unsigned short sW2T[64 * 64];   // W2^T, bf16
  __shared__ float sb1[64], sb2[64], sW1l[64], swe[64];
  __shared__ float sbe;
  __shared__ __align__(16) unsigned short sA [2][32 * 128]; // per-wave A tile (32 edges x 128 K)
  __shared__ __align__(16) unsigned short sM1[2][32 * 64];  // per-wave layer-1 output, bf16
  __shared__ float sSqd[2][32];
  __shared__ int   sDst[2][32];

  const int tid = threadIdx.x;
  // Stage transposed bf16 weights once per block (L2-resident, amortized).
  for (int i = tid; i < 64 * 128; i += 64) {
    int n = i >> 7, k = i & 127;
    sW1T[i] = f2bf(W1[k * 64 + n]);
  }
  for (int i = tid; i < 64 * 64; i += 64) {
    int n = i >> 6, k = i & 63;
    sW2T[i] = f2bf(W2[k * 64 + n]);
  }
  if (tid < 64) {
    sb1[tid]  = b1[tid];
    sb2[tid]  = b2[tid];
    sW1l[tid] = W1[128 * 64 + tid];   // last row of W1 (sqd column), kept in f32
    swe[tid]  = we[tid];
  }
  if (tid == 0) sbe = be[0];
  __syncthreads();

  const int wv = tid >> 5, lane = tid & 31;
  const int nlo = lane & 15, hi = lane >> 4;
  const int tile = blockIdx.x * 2 + wv;      // grid sized exactly: always valid
  const int e0 = tile * 32;

  // ---- gather: lane = edge index, loads src row -> cols 0..63, dst row -> 64..127
  {
    const int eg = e0 + lane;
    const int s = src[eg], d = dst[eg];
    const float4* sp = reinterpret_cast<const float4*>(feat + s * 64);
    const float4* dp = reinterpret_cast<const float4*>(feat + d * 64);
    uint2* ar = reinterpret_cast<uint2*>(sA[wv] + lane * 128);
#pragma unroll
    for (int j = 0; j < 16; ++j) {
      float4 f = sp[j];
      ar[j] = make_uint2(pk2(f.x, f.y), pk2(f.z, f.w));
    }
#pragma unroll
    for (int j = 0; j < 16; ++j) {
      float4 f = dp[j];
      ar[16 + j] = make_uint2(pk2(f.x, f.y), pk2(f.z, f.w));
    }
    // squared distance between endpoint coordinates
    float dx = x[s * 3 + 0] - x[d * 3 + 0];
    float dy = x[s * 3 + 1] - x[d * 3 + 1];
    float dz = x[s * 3 + 2] - x[d * 3 + 2];
    sSqd[wv][lane] = dx * dx + dy * dy + dz * dz;
    sDst[wv][lane] = d;
  }

  // ---- layer 1: [32 x 128] @ [128 x 64], 4 K-steps x 4 N-tiles x 2 M-tiles
  v8f acc[2][4];
#pragma unroll
  for (int nb = 0; nb < 4; ++nb) {
    float bv = sb1[nb * 16 + nlo];
    acc[0][nb] = (v8f){bv, bv, bv, bv, bv, bv, bv, bv};
    acc[1][nb] = acc[0][nb];
  }
#pragma unroll
  for (int kb = 0; kb < 4; ++kb) {
    const unsigned short* a0r = sA[wv] + nlo * 128 + kb * 32;
    const unsigned short* a1r = sA[wv] + (16 + nlo) * 128 + kb * 32;
    v16bf a0 = ld_frag(a0r + (hi ? 8 : 0), a0r + (hi ? 24 : 16));
    v16bf a1 = ld_frag(a1r + (hi ? 8 : 0), a1r + (hi ? 24 : 16));
#pragma unroll
    for (int nb = 0; nb < 4; ++nb) {
      const unsigned short* brow = sW1T + (nb * 16 + nlo) * 128 + kb * 32 + (hi ? 8 : 0);
      v16bf b = ld_frag(brow, brow + 16);
      acc[0][nb] = WMMA_BF16(a0, b, acc[0][nb]);   // independent accumulators:
      acc[1][nb] = WMMA_BF16(a1, b, acc[1][nb]);   // back-to-back issue, B reused
    }
  }
  // rank-1 update for the sqd column (K=128), relu, stash bf16 for layer 2
#pragma unroll
  for (int t = 0; t < 2; ++t)
#pragma unroll
    for (int nb = 0; nb < 4; ++nb) {
      float wl = sW1l[nb * 16 + nlo];
#pragma unroll
      for (int i = 0; i < 8; ++i) {
        int M = t * 16 + i + hi * 8;
        float v = acc[t][nb][i] + sSqd[wv][M] * wl;
        v = v > 0.f ? v : 0.f;
        sM1[wv][M * 64 + nb * 16 + nlo] = f2bf(v);
      }
    }

  // ---- layer 2: [32 x 64] @ [64 x 64]
  v8f acc2[2][4];
#pragma unroll
  for (int nb = 0; nb < 4; ++nb) {
    float bv = sb2[nb * 16 + nlo];
    acc2[0][nb] = (v8f){bv, bv, bv, bv, bv, bv, bv, bv};
    acc2[1][nb] = acc2[0][nb];
  }
#pragma unroll
  for (int kb = 0; kb < 2; ++kb) {
    const unsigned short* a0r = sM1[wv] + nlo * 64 + kb * 32;
    const unsigned short* a1r = sM1[wv] + (16 + nlo) * 64 + kb * 32;
    v16bf a0 = ld_frag(a0r + (hi ? 8 : 0), a0r + (hi ? 24 : 16));
    v16bf a1 = ld_frag(a1r + (hi ? 8 : 0), a1r + (hi ? 24 : 16));
#pragma unroll
    for (int nb = 0; nb < 4; ++nb) {
      const unsigned short* brow = sW2T + (nb * 16 + nlo) * 64 + kb * 32 + (hi ? 8 : 0);
      v16bf b = ld_frag(brow, brow + 16);
      acc2[0][nb] = WMMA_BF16(a0, b, acc2[0][nb]);
      acc2[1][nb] = WMMA_BF16(a1, b, acc2[1][nb]);
    }
  }
#pragma unroll
  for (int t = 0; t < 2; ++t)
#pragma unroll
    for (int nb = 0; nb < 4; ++nb)
#pragma unroll
      for (int i = 0; i < 8; ++i)
        acc2[t][nb][i] = acc2[t][nb][i] > 0.f ? acc2[t][nb][i] : 0.f;

  // ---- gate + scatter per M-tile
#pragma unroll
  for (int t = 0; t < 2; ++t) {
    // gate = sigmoid(m @ we + be): per-row dot via 16-lane xor reduction
    float gp[8];
#pragma unroll
    for (int i = 0; i < 8; ++i) gp[i] = 0.f;
#pragma unroll
    for (int nb = 0; nb < 4; ++nb) {
      float w = swe[nb * 16 + nlo];
#pragma unroll
      for (int i = 0; i < 8; ++i) gp[i] += acc2[t][nb][i] * w;
    }
#pragma unroll
    for (int m = 1; m < 16; m <<= 1)
#pragma unroll
      for (int i = 0; i < 8; ++i) gp[i] += __shfl_xor(gp[i], m, 32);

#pragma unroll
    for (int i = 0; i < 8; ++i) {
      const int M = t * 16 + i + hi * 8;
      const int dn = sDst[wv][M];
      const float g = 1.f / (1.f + expf(-(gp[i] + sbe)));
      float* base = msum + dn * 64 + nlo;
#pragma unroll
      for (int nb = 0; nb < 4; ++nb)
        unsafeAtomicAdd(base + nb * 16, acc2[t][nb][i] * g);
    }
  }
}

// ---------------------------------------------------------------------------
// Node kernel: out = relu((msum+feat)@U1+c1)@U2 + c2 + feat, 16-row tiles
// ---------------------------------------------------------------------------
__global__ __launch_bounds__(128) void egc_node_kernel(
    const float* __restrict__ feat, const float* __restrict__ msum,
    const float* __restrict__ U1, const float* __restrict__ c1,
    const float* __restrict__ U2, const float* __restrict__ c2,
    float* __restrict__ out)
{
  __shared__ __align__(16) unsigned short sU1T[64 * 64];
  __shared__ __align__(16) unsigned short sU2T[64 * 64];
  __shared__ float sc1[64], sc2[64];
  __shared__ __align__(16) unsigned short sH [4][16 * 64];
  __shared__ __align__(16) unsigned short sH2[4][16 * 64];
  __shared__ __align__(16) float          sF [4][16 * 64]; // feat tile, then output tile

  const int tid = threadIdx.x;
  for (int i = tid; i < 64 * 64; i += 128) {
    int n = i >> 6, k = i & 63;
    sU1T[i] = f2bf(U1[k * 64 + n]);
    sU2T[i] = f2bf(U2[k * 64 + n]);
  }
  if (tid < 64) { sc1[tid] = c1[tid]; sc2[tid] = c2[tid]; }
  __syncthreads();

  const int wv = tid >> 5, lane = tid & 31;
  const int nlo = lane & 15, hi = lane >> 4;
  const int tile = blockIdx.x * 4 + wv;
  if (tile >= (NN / 16)) return;             // wave-uniform guard: EXEC stays all-ones
  const int n0 = tile * 16;

  // stage h = msum + feat (bf16) and feat (f32), coalesced float4 loads
  {
    const int el = lane >> 1, half = lane & 1, col = half * 32;
    const float4* mp = reinterpret_cast<const float4*>(msum + (n0 + el) * 64 + col);
    const float4* fp = reinterpret_cast<const float4*>(feat + (n0 + el) * 64 + col);
    uint2* hr = reinterpret_cast<uint2*>(sH[wv] + el * 64 + col);
#pragma unroll
    for (int j = 0; j < 8; ++j) {
      float4 m = mp[j], f = fp[j];
      hr[j] = make_uint2(pk2(m.x + f.x, m.y + f.y), pk2(m.z + f.z, m.w + f.w));
      *reinterpret_cast<float4*>(sF[wv] + el * 64 + col + j * 4) = f;
    }
  }

  // layer 1: relu(h @ U1 + c1)
  v8f acc[4];
#pragma unroll
  for (int nb = 0; nb < 4; ++nb) {
    float bv = sc1[nb * 16 + nlo];
    acc[nb] = (v8f){bv, bv, bv, bv, bv, bv, bv, bv};
  }
#pragma unroll
  for (int kb = 0; kb < 2; ++kb) {
    const unsigned short* arow = sH[wv] + nlo * 64 + kb * 32;
    v16bf a = ld_frag(arow + (hi ? 8 : 0), arow + (hi ? 24 : 16));
#pragma unroll
    for (int nb = 0; nb < 4; ++nb) {
      const unsigned short* brow = sU1T + (nb * 16 + nlo) * 64 + kb * 32 + (hi ? 8 : 0);
      v16bf b = ld_frag(brow, brow + 16);
      acc[nb] = WMMA_BF16(a, b, acc[nb]);
    }
  }
#pragma unroll
  for (int nb = 0; nb < 4; ++nb)
#pragma unroll
    for (int i = 0; i < 8; ++i) {
      int M = i + hi * 8;
      float v = acc[nb][i];
      v = v > 0.f ? v : 0.f;
      sH2[wv][M * 64 + nb * 16 + nlo] = f2bf(v);
    }

  // layer 2: (...) @ U2 + c2
  v8f acc2[4];
#pragma unroll
  for (int nb = 0; nb < 4; ++nb) {
    float bv = sc2[nb * 16 + nlo];
    acc2[nb] = (v8f){bv, bv, bv, bv, bv, bv, bv, bv};
  }
#pragma unroll
  for (int kb = 0; kb < 2; ++kb) {
    const unsigned short* arow = sH2[wv] + nlo * 64 + kb * 32;
    v16bf a = ld_frag(arow + (hi ? 8 : 0), arow + (hi ? 24 : 16));
#pragma unroll
    for (int nb = 0; nb < 4; ++nb) {
      const unsigned short* brow = sU2T + (nb * 16 + nlo) * 64 + kb * 32 + (hi ? 8 : 0);
      v16bf b = ld_frag(brow, brow + 16);
      acc2[nb] = WMMA_BF16(a, b, acc2[nb]);
    }
  }

  // residual add into the staged feat tile, then coalesced float4 store
#pragma unroll
  for (int nb = 0; nb < 4; ++nb)
#pragma unroll
    for (int i = 0; i < 8; ++i) {
      int M = i + hi * 8, N = nb * 16 + nlo;
      sF[wv][M * 64 + N] += acc2[nb][i];
    }
  {
    const int el = lane >> 1, half = lane & 1, col = half * 32;
    float4* op = reinterpret_cast<float4*>(out + (n0 + el) * 64 + col);
    const float4* sp = reinterpret_cast<const float4*>(sF[wv] + el * 64 + col);
#pragma unroll
    for (int j = 0; j < 8; ++j) op[j] = sp[j];
  }
}

extern "C" void kernel_launch(void* const* d_in, const int* in_sizes, int n_in,
                              void* d_out, int out_size, void* d_ws, size_t ws_size,
                              hipStream_t stream) {
  (void)in_sizes; (void)n_in; (void)out_size; (void)ws_size;
  const float* feat = (const float*)d_in[0];
  const float* x    = (const float*)d_in[1];
  const int*   src  = (const int*)d_in[2];
  const int*   dst  = (const int*)d_in[3];
  const float* W1 = (const float*)d_in[4];
  const float* b1 = (const float*)d_in[5];
  const float* W2 = (const float*)d_in[6];
  const float* b2 = (const float*)d_in[7];
  const float* we = (const float*)d_in[8];
  const float* be = (const float*)d_in[9];
  const float* U1 = (const float*)d_in[10];
  const float* c1 = (const float*)d_in[11];
  const float* U2 = (const float*)d_in[12];
  const float* c2 = (const float*)d_in[13];
  float* out  = (float*)d_out;
  float* msum = (float*)d_ws;            // [NN, 64] f32 accumulator

  hipMemsetAsync(msum, 0, (size_t)NN * HH * sizeof(float), stream);
  // 800000 edges / 32 per wave / 2 waves per block = 12500 blocks (exact)
  egc_edge_kernel<<<EE / 32 / 2, 64, 0, stream>>>(feat, x, src, dst,
                                                  W1, b1, W2, b2, we, be, msum);
  // 50000 nodes / 16 per wave = 3125 tiles -> 782 blocks of 4 waves
  egc_node_kernel<<<(NN / 16 + 3) / 4, 128, 0, stream>>>(feat, msum, U1, c1, U2, c2, out);
}
